// FastAttention_91336774517462
// MI455X (gfx1250) — matrix operand order repaired and checked by reading, not compiled, not run on hardware
//
#include <hip/hip_runtime.h>
#include <hip/hip_bf16.h>

// Performer (FAVOR+) non-causal attention for MI455X (gfx1250, wave32, WMMA).
// B=4,H=8 -> BH=32 heads; N=8192 rows; D=64 head dim; M=256 features.

#define BH 32
#define NN 8192
#define DD 64
#define MM 256

static constexpr float DN    = 0.35355339059327379f;  // 64^-0.25
static constexpr float RATIO = 0.0625f;               // 256^-0.5
static constexpr float KEPS  = 1e-4f;
// diag factor: 0.5 * DN^2 = 0.5/8 = 0.0625

// padded LDS strides (multiples of 8 halves -> 16B-aligned b128 DS ops)
#define VT_STRIDE  136   // transposed V   [64 e][128 rows + pad]
#define CT_STRIDE  264   // transposed ctx [64 e][256 m + pad]
#define BN_STRIDE  40    // bounce scratch row stride

typedef __attribute__((ext_vector_type(16))) _Float16 v16h;
typedef __attribute__((ext_vector_type(8)))  _Float16 v8h;
typedef __attribute__((ext_vector_type(8)))  float    v8f;

__device__ __forceinline__ int laneid() { return threadIdx.x & 31; }

__device__ __forceinline__ v8f zero8() {
  v8f z = {0.f,0.f,0.f,0.f,0.f,0.f,0.f,0.f};
  return z;
}

__device__ __forceinline__ v8f wmma16(v16h a, v16h b, v8f c) {
  return __builtin_amdgcn_wmma_f32_16x16x32_f16(false, a, false, b, (short)0, c, false, false);
}

// ----- single-instruction wave32 lane exchange via ds_swizzle -----
// group-of-32 mode: src_lane = ((lane & and) | or) ^ xor
template <int OFF>
__device__ __forceinline__ float swzf(float x) {
  return __int_as_float(__builtin_amdgcn_ds_swizzle(__float_as_int(x), OFF));
}
#define SWZ_XOR(m)   (((m) << 10) | 0x1f)   // butterfly xor within wave32
#define SWZ_BCAST16  (0x10)                 // src = lane & 16 (lanes 0 / 16)

// ----- ordered-uint encoding for float atomicMax -----
__device__ __forceinline__ unsigned f2ord(float f) {
  unsigned u = __float_as_uint(f);
  return (u & 0x80000000u) ? ~u : (u | 0x80000000u);
}
__device__ __forceinline__ float ord2f(unsigned e) {
  unsigned u = (e & 0x80000000u) ? (e & 0x7fffffffu) : ~e;
  return __uint_as_float(u);
}

// ----- fragment builders (contiguous k = 16*(lane/16) + s for A and B, wave32) -----

// A fragment 16x32 f16 from LDS tile [rows][stride halves], row-major. 2x ds_load_b128.
__device__ __forceinline__ v16h afrag_lds(const _Float16* lds, int row0, int k0, int stride) {
  const int l = laneid();
  const int m  = l & 15;
  const int kb = (l >> 4) << 4;   // 0 or 16
  const _Float16* p = lds + (row0 + m) * stride + k0 + kb;
  v8h lo = *(const v8h*)p;        // K = k0+kb .. +7
  v8h hi = *(const v8h*)(p + 8);  // K = k0+kb+8 .. +7
  v16h r;
#pragma unroll
  for (int i = 0; i < 8; ++i) { r[i] = lo[i]; r[i + 8] = hi[i]; }
  return r;
}

// B fragment 32x16: B[k][n] = proj[col0+n][k0+k] (proj row-major [256][64], f32 global).
// Per lane: 16 consecutive floats -> coalesced per-lane b128 loads + v_cvt_pk.
__device__ __forceinline__ v16h bfrag_projT_global(const float* __restrict__ proj, int col0, int k0) {
  const int l = laneid();
  const int n  = l & 15;
  const int kb = (l >> 4) << 4;   // 0 or 16
  const float* p = proj + (col0 + n) * DD + k0 + kb;
  v16h r;
#pragma unroll
  for (int i = 0; i < 16; ++i) r[i] = (_Float16)p[i];
  return r;
}

// B fragment 32x16 from TRANSPOSED LDS: element (k,n) at ldsT[(n0+n)*stride + k0+k].
// 16 consecutive halves per lane -> 2x ds_load_b128.
__device__ __forceinline__ v16h bfrag_ldsT(const _Float16* ldsT, int k0, int n0, int stride) {
  const int l = laneid();
  const int n  = l & 15;
  const int kb = (l >> 4) << 4;   // 0 or 16
  const _Float16* p = ldsT + (n0 + n) * stride + k0 + kb;
  v8h lo = *(const v8h*)p;
  v8h hi = *(const v8h*)(p + 8);
  v16h r;
#pragma unroll
  for (int i = 0; i < 8; ++i) { r[i] = lo[i]; r[i + 8] = hi[i]; }
  return r;
}

// B fragment with ones in column 0 (for row-sum via WMMA).
__device__ __forceinline__ v16h bfrag_ones_col0() {
  const _Float16 v = ((laneid() & 15) == 0) ? (_Float16)1.0f : (_Float16)0.0f;
  v16h r;
#pragma unroll
  for (int i = 0; i < 16; ++i) r[i] = v;
  return r;
}

// B fragment with ksum in column 0 (from f32 LDS array).
__device__ __forceinline__ v16h bfrag_col0_f32(const float* lds, int k0) {
  const int l = laneid();
  const int kb = (l >> 4) << 4;
  const bool on = ((l & 15) == 0);
  v16h r;
#pragma unroll
  for (int i = 0; i < 16; ++i) {
    float x = lds[k0 + kb + i];
    r[i] = on ? (_Float16)x : (_Float16)0.0f;
  }
  return r;
}

// A fragment [feat16 x row32] from bounce scratch laid out [16 feats][BN_STRIDE rows].
// 2x ds_load_b128 per lane.
__device__ __forceinline__ v16h a2frag_T(const _Float16* s) {
  const int l = laneid();
  const int f  = l & 15;
  const int kb = (l >> 4) << 4;   // 0 or 16
  const _Float16* p = s + f * BN_STRIDE + kb;
  v8h lo = *(const v8h*)p;         // rows kb..kb+7
  v8h hi = *(const v8h*)(p + 8);   // rows kb+8..+7
  v16h r;
#pragma unroll
  for (int i = 0; i < 8; ++i) { r[i] = lo[i]; r[i + 8] = hi[i]; }
  return r;
}

// A fragment [row16 x feat32] from bounce scratch laid out [16 rows][BN_STRIDE feats].
__device__ __forceinline__ v16h afrag_scratch32(const _Float16* s) {
  const int l = laneid();
  const int row = l & 15;
  const int kb  = (l >> 4) << 4;
  const _Float16* p = s + row * BN_STRIDE + kb;
  v8h lo = *(const v8h*)p;
  v8h hi = *(const v8h*)(p + 8);
  v16h r;
#pragma unroll
  for (int i = 0; i < 8; ++i) { r[i] = lo[i]; r[i + 8] = hi[i]; }
  return r;
}

// =====================================================================
// Kernel 0: init workspace (kmax = -inf encoded, ksum/context = 0)
// =====================================================================
__global__ void k_init(unsigned* __restrict__ kmax, float* __restrict__ ksum,
                       float* __restrict__ ctx) {
  const long tid = (long)blockIdx.x * blockDim.x + threadIdx.x;
  const long nks = (long)BH * MM;
  const long nctx = (long)BH * MM * DD;
  for (long i = tid; i < nks + nctx; i += (long)gridDim.x * blockDim.x) {
    if (i < nks) ksum[i] = 0.f;
    else         ctx[i - nks] = 0.f;
  }
  if (tid < BH) kmax[tid] = f2ord(-3.0e38f);
}

// =====================================================================
// Kernel 1: global per-head max of k_dash = (DN*k) @ proj^T
// grid (64 chunks, 32 heads), 256 threads = 8 waves, 128 rows/block
// =====================================================================
__global__ void __launch_bounds__(256) k_kmax(const float* __restrict__ K,
                                              const float* __restrict__ proj,
                                              unsigned* __restrict__ kmax) {
  __shared__ __align__(16) _Float16 sK[128 * DD];
  const int bh = blockIdx.y;
  const long gbase = (long)bh * NN * DD + (long)blockIdx.x * 128 * DD;
  for (int i = threadIdx.x; i < 128 * DD; i += 256)
    sK[i] = (_Float16)(K[gbase + i] * DN);
  __syncthreads();

  const int w = threadIdx.x >> 5;
  v16h a0 = afrag_lds(sK, w * 16, 0, DD);
  v16h a1 = afrag_lds(sK, w * 16, 32, DD);
  float mx = -3.0e38f;
  for (int mt = 0; mt < 16; ++mt) {
    v16h b0 = bfrag_projT_global(proj, mt * 16, 0);
    v16h b1 = bfrag_projT_global(proj, mt * 16, 32);
    v8f acc = zero8();
    acc = wmma16(a0, b0, acc);
    acc = wmma16(a1, b1, acc);
#pragma unroll
    for (int r = 0; r < 8; ++r) mx = fmaxf(mx, acc[r]);
  }
  mx = fmaxf(mx, swzf<SWZ_XOR(1)>(mx));
  mx = fmaxf(mx, swzf<SWZ_XOR(2)>(mx));
  mx = fmaxf(mx, swzf<SWZ_XOR(4)>(mx));
  mx = fmaxf(mx, swzf<SWZ_XOR(8)>(mx));
  mx = fmaxf(mx, swzf<SWZ_XOR(16)>(mx));
  if (laneid() == 0) atomicMax(&kmax[bh], f2ord(mx));
}

// =====================================================================
// Kernel 2: kp = ratio*(exp(k_dash - diag - kmax)+eps);
//           context[m][e] += kp^T v ; ksum[m] += sum_n kp
// grid (16, 32); each block: 512 rows in 4 chunks of 128.
// Wave w owns feature tiles {2w, 2w+1}; proj B-fragments register-resident.
// =====================================================================
__global__ void __launch_bounds__(256) k_ctx(const float* __restrict__ K,
                                             const float* __restrict__ V,
                                             const float* __restrict__ proj,
                                             const unsigned* __restrict__ kmaxbuf,
                                             float* __restrict__ ksum,
                                             float* __restrict__ ctx) {
  __shared__ __align__(16) _Float16 sK[128 * DD];
  __shared__ __align__(16) _Float16 sVT[DD * VT_STRIDE];         // transposed [e][row]
  __shared__ float    sDiag[128];
  __shared__ __align__(16) _Float16 sBounce[8][16 * BN_STRIDE];  // [feat][row]

  const int bh = blockIdx.y;
  const float kmx = ord2f(kmaxbuf[bh]);
  const int w  = threadIdx.x >> 5;
  const int l  = laneid();
  const int hl = l >> 4;
  const int mt0 = w * 2;

  v16h bp00 = bfrag_projT_global(proj, mt0 * 16, 0);
  v16h bp01 = bfrag_projT_global(proj, mt0 * 16, 32);
  v16h bp10 = bfrag_projT_global(proj, (mt0 + 1) * 16, 0);
  v16h bp11 = bfrag_projT_global(proj, (mt0 + 1) * 16, 32);
  v16h bones = bfrag_ones_col0();

  v8f accCtx[2][4];
  v8f accKs[2];
#pragma unroll
  for (int mi = 0; mi < 2; ++mi) {
    accKs[mi] = zero8();
#pragma unroll
    for (int et = 0; et < 4; ++et) accCtx[mi][et] = zero8();
  }

  for (int c = 0; c < 4; ++c) {
    const long gbase = (long)bh * NN * DD + ((long)blockIdx.x * 512 + c * 128) * DD;
    for (int i = threadIdx.x; i < 128 * DD; i += 256) {
      sK[i] = (_Float16)(K[gbase + i] * DN);
      const int row = i >> 6, e = i & 63;
      sVT[e * VT_STRIDE + row] = (_Float16)(V[gbase + i]);
    }
    {  // diag[row] = 0.0625 * sum(k_raw^2)
      const int row = threadIdx.x >> 1, hh = threadIdx.x & 1;
      const float* p = K + gbase + row * DD + hh * 32;
      float s = 0.f;
#pragma unroll
      for (int i = 0; i < 32; ++i) { float x = p[i]; s += x * x; }
      s += swzf<SWZ_XOR(1)>(s);
      if (!hh) sDiag[row] = 0.0625f * s;
    }
    if (c < 3) {  // prefetch next chunk (global_prefetch_b8, one 128B line/thread)
      const float* nk = K + gbase + 128 * DD + (long)threadIdx.x * 32;
      const float* nv = V + gbase + 128 * DD + (long)threadIdx.x * 32;
      __builtin_prefetch(nk, 0, 0);
      __builtin_prefetch(nv, 0, 0);
    }
    __syncthreads();

    for (int st = 0; st < 4; ++st) {
      const int rs = st * 32;
      v16h a00 = afrag_lds(sK, rs,      0,  DD);
      v16h a01 = afrag_lds(sK, rs,      32, DD);
      v16h a10 = afrag_lds(sK, rs + 16, 0,  DD);
      v16h a11 = afrag_lds(sK, rs + 16, 32, DD);
      v16h bv0 = bfrag_ldsT(sVT, rs, 0,  VT_STRIDE);
      v16h bv1 = bfrag_ldsT(sVT, rs, 16, VT_STRIDE);
      v16h bv2 = bfrag_ldsT(sVT, rs, 32, VT_STRIDE);
      v16h bv3 = bfrag_ldsT(sVT, rs, 48, VT_STRIDE);
#pragma unroll
      for (int mi = 0; mi < 2; ++mi) {
        v8f d0 = zero8(), d1 = zero8();
        d0 = wmma16(a00, mi ? bp10 : bp00, d0);
        d0 = wmma16(a01, mi ? bp11 : bp01, d0);
        d1 = wmma16(a10, mi ? bp10 : bp00, d1);
        d1 = wmma16(a11, mi ? bp11 : bp01, d1);
        // kp + transpose bounce: pack 8 rows/lane -> 2x ds_store_b128
        v8h kp0, kp1;
#pragma unroll
        for (int r = 0; r < 8; ++r) {
          const int r0 = r + 8 * hl;
          kp0[r] = (_Float16)(RATIO * (__expf(d0[r] - sDiag[rs + r0] - kmx) + KEPS));
          kp1[r] = (_Float16)(RATIO * (__expf(d1[r] - sDiag[rs + 16 + r0] - kmx) + KEPS));
        }
        _Float16* bb = &sBounce[w][(l & 15) * BN_STRIDE + 8 * hl];
        *(v8h*)bb        = kp0;   // rows 8hl..8hl+7
        *(v8h*)(bb + 16) = kp1;   // rows 16+8hl..+7
        asm volatile("s_wait_dscnt 0" ::: "memory");  // intra-wave LDS transpose bounce
        v16h a2 = a2frag_T(&sBounce[w][0]);
        accKs[mi]     = wmma16(a2, bones, accKs[mi]);
        accCtx[mi][0] = wmma16(a2, bv0, accCtx[mi][0]);
        accCtx[mi][1] = wmma16(a2, bv1, accCtx[mi][1]);
        accCtx[mi][2] = wmma16(a2, bv2, accCtx[mi][2]);
        accCtx[mi][3] = wmma16(a2, bv3, accCtx[mi][3]);
      }
    }
    __syncthreads();
  }

  // flush partials
#pragma unroll
  for (int mi = 0; mi < 2; ++mi) {
#pragma unroll
    for (int r = 0; r < 8; ++r) {
      const int f = (mt0 + mi) * 16 + r + 8 * hl;
      if ((l & 15) == 0) atomicAdd(&ksum[bh * MM + f], accKs[mi][r]);
#pragma unroll
      for (int et = 0; et < 4; ++et) {
        const int e = et * 16 + (l & 15);
        atomicAdd(&ctx[((long)bh * MM + f) * DD + e], accCtx[mi][et][r]);
      }
    }
  }
}

// =====================================================================
// Kernel 3: qp (per-row max), denominator qp.ksum, out = (qp@ctx)/den
// grid (64, 32); 128 rows/block; wave owns 16 rows, q_dash in registers.
// =====================================================================
__global__ void __launch_bounds__(256) k_out(const float* __restrict__ Q,
                                             const float* __restrict__ proj,
                                             const float* __restrict__ ksum,
                                             const float* __restrict__ ctx,
                                             float* __restrict__ out) {
  __shared__ __align__(16) _Float16 sQ[128 * DD];
  __shared__ float    sDiag[128];
  __shared__ __align__(16) _Float16 sCtxT[DD * CT_STRIDE];       // transposed [e][m]
  __shared__ float    sKs[MM];
  __shared__ __align__(16) _Float16 sBounce[8][16 * BN_STRIDE];  // [row][feat]

  const int bh = blockIdx.y;
  const long rowsBase = (long)blockIdx.x * 128;
  const long gbase = (long)bh * NN * DD + rowsBase * DD;

  for (int i = threadIdx.x; i < 128 * DD; i += 256)
    sQ[i] = (_Float16)(Q[gbase + i] * DN);
  for (int i = threadIdx.x; i < MM * DD; i += 256) {
    const int m = i >> 6, e = i & 63;
    sCtxT[e * CT_STRIDE + m] = (_Float16)(ctx[(long)bh * MM * DD + i]);
  }
  sKs[threadIdx.x] = ksum[bh * MM + threadIdx.x];
  {
    const int row = threadIdx.x >> 1, hh = threadIdx.x & 1;
    const float* p = Q + gbase + row * DD + hh * 32;
    float s = 0.f;
#pragma unroll
    for (int i = 0; i < 32; ++i) { float x = p[i]; s += x * x; }
    s += swzf<SWZ_XOR(1)>(s);
    if (!hh) sDiag[row] = 0.0625f * s;
  }
  __syncthreads();

  const int w  = threadIdx.x >> 5;
  const int l  = laneid();
  const int hl = l >> 4;
  const int row0 = w * 16;

  v16h aq0 = afrag_lds(sQ, row0, 0, DD);
  v16h aq1 = afrag_lds(sQ, row0, 32, DD);

  v8f qd[16];
#pragma unroll
  for (int mt = 0; mt < 16; ++mt) {
    v16h b0 = bfrag_projT_global(proj, mt * 16, 0);
    v16h b1 = bfrag_projT_global(proj, mt * 16, 32);
    v8f acc = zero8();
    acc = wmma16(aq0, b0, acc);
    qd[mt] = wmma16(aq1, b1, acc);
  }

  // per-row max (N striped over 16-lane halves), diag, exp
  float dg[8], rm[8];
#pragma unroll
  for (int r = 0; r < 8; ++r) {
    dg[r] = sDiag[row0 + r + 8 * hl];
    float m = qd[0][r];
#pragma unroll
    for (int mt = 1; mt < 16; ++mt) m = fmaxf(m, qd[mt][r]);
    m = fmaxf(m, swzf<SWZ_XOR(1)>(m));
    m = fmaxf(m, swzf<SWZ_XOR(2)>(m));
    m = fmaxf(m, swzf<SWZ_XOR(4)>(m));
    m = fmaxf(m, swzf<SWZ_XOR(8)>(m));
    rm[r] = m;
  }
#pragma unroll
  for (int mt = 0; mt < 16; ++mt)
#pragma unroll
    for (int r = 0; r < 8; ++r)
      qd[mt][r] = RATIO * (__expf(qd[mt][r] - dg[r] - rm[r]) + KEPS);

  // out = qp @ ctx ; den = qp . ksum (column-0 fragment)
  v8f accO0 = zero8(), accO1 = zero8(), accO2 = zero8(), accO3 = zero8();
  v8f accD = zero8();
#pragma unroll
  for (int ks = 0; ks < 8; ++ks) {
#pragma unroll
    for (int t = 0; t < 2; ++t)
#pragma unroll
      for (int r = 0; r < 8; ++r)
        sBounce[w][(r + 8 * hl) * BN_STRIDE + t * 16 + (l & 15)] = (_Float16)qd[2 * ks + t][r];
    asm volatile("s_wait_dscnt 0" ::: "memory");
    v16h a2 = afrag_scratch32(&sBounce[w][0]);
    v16h bks = bfrag_col0_f32(sKs, ks * 32);
    accD = wmma16(a2, bks, accD);
    accO0 = wmma16(a2, bfrag_ldsT(sCtxT, ks * 32, 0,  CT_STRIDE), accO0);
    accO1 = wmma16(a2, bfrag_ldsT(sCtxT, ks * 32, 16, CT_STRIDE), accO1);
    accO2 = wmma16(a2, bfrag_ldsT(sCtxT, ks * 32, 32, CT_STRIDE), accO2);
    accO3 = wmma16(a2, bfrag_ldsT(sCtxT, ks * 32, 48, CT_STRIDE), accO3);
  }

#pragma unroll
  for (int r = 0; r < 8; ++r) {
    const float den = swzf<SWZ_BCAST16>(accD[r]);  // col 0 lives in lanes 0/16
    const float inv = 1.0f / den;
    const long row = rowsBase + row0 + r + 8 * hl;
    float* op = out + ((long)bh * NN + row) * DD + (l & 15);
    op[0]  = accO0[r] * inv;
    op[16] = accO1[r] * inv;
    op[32] = accO2[r] * inv;
    op[48] = accO3[r] * inv;
  }
}

// =====================================================================
extern "C" void kernel_launch(void* const* d_in, const int* in_sizes, int n_in,
                              void* d_out, int out_size, void* d_ws, size_t ws_size,
                              hipStream_t stream) {
  const float* q    = (const float*)d_in[0];
  const float* k    = (const float*)d_in[1];
  const float* v    = (const float*)d_in[2];
  const float* proj = (const float*)d_in[3];
  float* out = (float*)d_out;

  unsigned* kmax = (unsigned*)d_ws;                       // 32 slots
  float*    ksum = (float*)((char*)d_ws + 256);           // 32*256 f32
  float*    ctx  = ksum + (long)BH * MM;                  // 32*256*64 f32

  k_init<<<dim3(512), dim3(256), 0, stream>>>(kmax, ksum, ctx);
  k_kmax<<<dim3(64, 32), dim3(256), 0, stream>>>(k, proj, kmax);
  k_ctx <<<dim3(16, 32), dim3(256), 0, stream>>>(k, v, proj, kmax, ksum, ctx);
  k_out <<<dim3(64, 32), dim3(256), 0, stream>>>(q, proj, ksum, ctx, out);
}
